// NeighborhoodAttention_2190433321583
// MI455X (gfx1250) — matrix operand order repaired and checked by reading, not compiled, not run on hardware
//
#include <hip/hip_runtime.h>

// ---------------------------------------------------------------------------
// Neighborhood attention (ws=7) on gfx1250 / MI455X, wave32 + WMMA f16.
// Pipeline: [qkv_gemm] -> [na_attn (masked-union WMMA attention, async-LDS V
// staging)] -> [proj_gemm]
// Roofline: ~1.9 GFLOP vs ~tens of MB traffic @ 23.3 TB/s => memory bound;
// f16 intermediates + v_wmma_f32_16x16x32_f16 (f32 accumulate) everywhere.
// ---------------------------------------------------------------------------

typedef __attribute__((ext_vector_type(16))) _Float16 v16h;
typedef __attribute__((ext_vector_type(8)))  _Float16 v8h;
typedef __attribute__((ext_vector_type(8)))  float    v8f;

#define WSZ   7
#define PADW  3
#define NH    4
#define HD    16
#define CCH   64
#define WIMG  128
#define HWPX  16384   // 128*128

__device__ __forceinline__ int iclamp(int v, int lo, int hi) {
  return v < lo ? lo : (v > hi ? hi : v);
}
// bias row/col class: rows 0..2 -> 0..2, 3..124 -> 3, 125..127 -> 4..6
__device__ __forceinline__ int bclass(int i) {
  return i <= 2 ? i : (i >= 125 ? i - 121 : 3);
}
// 16-bit operand K order used here (contiguous): element e of the v16h held by a
// lane in half `hh` (0: lanes 0-15, 1: lanes 16-31) maps to K index 16*hh + e:
__device__ __forceinline__ int amapK(int e, int hh) {
  return 16 * hh + e;
}

// ---------------------------------------------------------------------------
// Kernel 1: qkv = einsum('bchw,oc->bohw', x, qkv_w) + qkv_b, split into
// q (pre-scaled by hd^-0.5), k, v stored f16 as [b][head][pixel][d].
// One wave per 16-pixel x 16-output tile; K = 64 channels = 2 WMMA steps.
// ---------------------------------------------------------------------------
__global__ __launch_bounds__(256) void qkv_gemm(
    const float* __restrict__ x, const float* __restrict__ w,
    const float* __restrict__ bias,
    _Float16* __restrict__ qb, _Float16* __restrict__ kb,
    _Float16* __restrict__ vb) {
  const int wid = threadIdx.x >> 5, lane = threadIdx.x & 31;
  const int hh = lane >> 4, lq = lane & 15;
  int gid = blockIdx.x * 8 + wid;          // 24576 waves total
  int mt = gid / 12, nt = gid - mt * 12;   // 2048 M-tiles x 12 N-tiles
  int pix = mt * 16 + lq;
  int b = pix >> 14, hw = pix & (HWPX - 1);
  const float* xb = x + (size_t)b * CCH * HWPX;

  v8f acc = {};
#pragma unroll
  for (int ks = 0; ks < 2; ++ks) {
    v16h a, bv;
#pragma unroll
    for (int e = 0; e < 16; ++e) {
      int c = ks * 32 + amapK(e, hh);                 // channel (K dim)
      a[e] = (_Float16)xb[(size_t)c * HWPX + hw];     // x is channel-major
    }
    int n = nt * 16 + lq;                             // output column
    const float4* wp = (const float4*)(w + (size_t)n * CCH + ks * 32 + hh * 16);
#pragma unroll
    for (int q = 0; q < 4; ++q) {
      float4 w4 = wp[q];
      bv[q * 4 + 0] = (_Float16)w4.x; bv[q * 4 + 1] = (_Float16)w4.y;
      bv[q * 4 + 2] = (_Float16)w4.z; bv[q * 4 + 3] = (_Float16)w4.w;
    }
    acc = __builtin_amdgcn_wmma_f32_16x16x32_f16(false, a, false, bv,
                                                 (short)0, acc, false, false);
  }
  // epilogue: D layout lane holds col o = nt*16+lq, rows m = j + 8*hh
  int o = nt * 16 + lq;
  float bia = bias[o];
  int sec = nt >> 2;        // 0=q, 1=k, 2=v  (uniform in wave)
  int head = nt & 3;        // uniform in wave
  _Float16* dst = (sec == 0) ? qb : (sec == 1) ? kb : vb;
  float scl = (sec == 0) ? 0.25f : 1.0f;   // hd^-0.5 = 1/4 folded into q
#pragma unroll
  for (int j = 0; j < 8; ++j) {
    int m = j + 8 * hh;
    int pj = mt * 16 + m;
    int bj = pj >> 14, hwj = pj & (HWPX - 1);
    dst[(((size_t)bj * NH + head) * HWPX + hwj) * HD + lq] =
        (_Float16)((acc[j] + bia) * scl);
  }
}

// ---------------------------------------------------------------------------
// Kernel 2: neighborhood attention. One wave per (batch, head, 4x4 query
// tile). Key union of the tile is <= 10x10 = 100 pixels -> up to 7 key tiles
// of 16. The V union is staged into LDS with global_load_async_to_lds_b128
// (ASYNCcnt-tracked, s_wait_asynccnt before use). QK^T: Q as A (K padded
// 16->32), keys as B. Window mask + relative bias folded into logits; row
// softmax via width-16 shuffles; probabilities staged f16 in LDS (stride 136
// halfs to avoid bank conflicts) and reloaded in A layout for AV.
// Divisions by nkc use an exact magic multiply (kk<=127, d<=10).
// ---------------------------------------------------------------------------
__global__ __launch_bounds__(128) void na_attn(
    const _Float16* __restrict__ qbuf, const _Float16* __restrict__ kbuf,
    const _Float16* __restrict__ vbuf, const float* __restrict__ rbias,
    _Float16* __restrict__ obuf) {
  __shared__ _Float16 attn_s[4][16 * 136];   // per-wave attn matrix
  __shared__ _Float16 v_s[4][128 * HD];      // per-wave staged V union
  const int wid = threadIdx.x >> 5, lane = threadIdx.x & 31;
  const int hh = lane >> 4, lq = lane & 15;
  _Float16* my = attn_s[wid];
  _Float16* vlds = v_s[wid];

  int gid = blockIdx.x * 4 + wid;   // 8192 waves: 2 batch * 1024 tiles * 4 heads
  int h = gid & 3;
  int tile = gid >> 2;
  int b = tile >> 10;
  int ty = (tile >> 5) & 31, tx = tile & 31;
  int qr0 = ty * 4, qc0 = tx * 4;
  int cr0 = iclamp(qr0, PADW, 124), cr1 = iclamp(qr0 + 3, PADW, 124);
  int cc0 = iclamp(qc0, PADW, 124), cc1 = iclamp(qc0 + 3, PADW, 124);
  int kr0 = cr0 - PADW, kc0 = cc0 - PADW;
  int nkr = cr1 - cr0 + WSZ, nkc = cc1 - cc0 + WSZ;   // 7..10 each
  int nkeys = nkr * nkc;                              // <= 100 (uniform)
  // exact magic divide by nkc for kk <= 127 (one uniform udiv per wave)
  unsigned mg = 65535u / (unsigned)nkc + 1u;

  int bh = b * NH + h;
  const _Float16* qB = qbuf + (size_t)bh * HWPX * HD;
  const _Float16* kB = kbuf + (size_t)bh * HWPX * HD;
  const _Float16* vB = vbuf + (size_t)bh * HWPX * HD;

  // ---- zero the tail of the V stage (slots k >= nkeys); these bytes are
  // never touched by the async engine, so the DS and ASYNC pipes (which are
  // mutually unordered) never overlap.
  {
    v8h zz = {};
#pragma unroll
    for (int i = 0; i < 4; ++i) {
      int k = lane + i * 32;
      if (k >= nkeys) {
        ((v8h*)(vlds + k * HD))[0] = zz;
        ((v8h*)(vlds + k * HD))[1] = zz;
      }
    }
  }
  // ---- async-stage V union into LDS: per lane, slots lane, lane+32, ...
  // Each slot is 32B = two b128 transfers (INST_OFFSET applies to both the
  // memory and LDS address, so offset:16 lands at vlds + k*32 + 16).
  {
    unsigned long long vbase = (unsigned long long)(size_t)vB;
#pragma unroll
    for (int i = 0; i < 4; ++i) {
      int k = lane + i * 32;
      if (k < nkeys) {
        unsigned q = ((unsigned)k * mg) >> 16;
        int kr = (int)q, kc = k - (int)q * nkc;
        unsigned moff = (unsigned)((((kr0 + kr) * WIMG) + (kc0 + kc)) * HD * 2);
        unsigned laddr = (unsigned)(size_t)(vlds + k * HD);
        asm volatile(
            "global_load_async_to_lds_b128 %0, %1, %2\n\t"
            "global_load_async_to_lds_b128 %0, %1, %2 offset:16"
            :
            : "v"(laddr), "v"(moff), "s"(vbase)
            : "memory");
      }
    }
  }

  // ---- zero this wave's LDS attn matrix (16 rows x 136 halfs = 272 v8h)
  {
    v8h zz = {};
    v8h* p = (v8h*)my;
#pragma unroll
    for (int i = 0; i < 9; ++i) {
      int idx = i * 32 + lane;
      if (idx < 272) p[idx] = zz;
    }
  }

  // ---- Q fragment: A row m = lq -> query (m>>2, m&3); d = e + 8*hh for e<8
  int qrow = qr0 + (lq >> 2), qcol = qc0 + (lq & 3);
  v16h qa = {};
  {
    const v8h* p = (const v8h*)(qB + ((size_t)qrow * WIMG + qcol) * HD + hh * 8);
    v8h t = *p;
#pragma unroll
    for (int e = 0; e < 8; ++e) qa[e] = t[e];
  }

  // ---- QK^T over up to 7 key tiles (guards are wave-uniform: EXEC stays 1s)
  v8f accs[7];
#pragma unroll
  for (int t = 0; t < 7; ++t) {
    if (t * 16 < nkeys) {
      v16h kv = {};
      int kk = t * 16 + lq;
      if (kk < nkeys) {   // B layout: half hh carries K(=d) 8*hh..8*hh+7 in elements 0..7
        unsigned q = ((unsigned)kk * mg) >> 16;
        int krow = kr0 + (int)q, kcol = kc0 + (kk - (int)q * nkc);
        const v8h* p = (const v8h*)(kB + ((size_t)krow * WIMG + kcol) * HD + hh * 8);
        v8h lo = p[0];
#pragma unroll
        for (int e = 0; e < 8; ++e) { kv[e] = lo[e]; }
      }
      v8f z = {};
      accs[t] = __builtin_amdgcn_wmma_f32_16x16x32_f16(false, qa, false, kv,
                                                       (short)0, z, false, false);
    }
  }

  // ---- window mask + relative bias (D layout: col = lq, row = j + 8*hh)
#pragma unroll
  for (int t = 0; t < 7; ++t) {
    if (t * 16 < nkeys) {
      int kk = t * 16 + lq;
      bool cv = kk < nkeys;
      unsigned q = ((unsigned)kk * mg) >> 16;
      int krow = kr0 + (int)q, kcol = kc0 + (kk - (int)q * nkc);
#pragma unroll
      for (int j = 0; j < 8; ++j) {
        int m = j + 8 * hh;
        int qr = qr0 + (m >> 2), qc = qc0 + (m & 3);
        int ci = iclamp(qr, PADW, 124), cj = iclamp(qc, PADW, 124);
        int di = krow - (ci - PADW), dj = kcol - (cj - PADW);
        bool ok = cv && ((unsigned)di < WSZ) && ((unsigned)dj < WSZ);
        float lv = -1e30f;
        if (ok) {
          int bi = (bclass(qr) + di) * (2 * WSZ - 1) + (bclass(qc) + dj);
          lv = accs[t][j] + rbias[bi * NH + h];
        }
        accs[t][j] = lv;
      }
    }
  }

  // ---- row softmax (row spread over 16 lanes of this half) + LDS store f16
#pragma unroll
  for (int j = 0; j < 8; ++j) {
    float mx = -1e30f;
#pragma unroll
    for (int t = 0; t < 7; ++t)
      if (t * 16 < nkeys) mx = fmaxf(mx, accs[t][j]);
#pragma unroll
    for (int s = 1; s < 16; s <<= 1) mx = fmaxf(mx, __shfl_xor(mx, s, 16));
    float sum = 0.f;
#pragma unroll
    for (int t = 0; t < 7; ++t)
      if (t * 16 < nkeys) {
        float pv = __expf(accs[t][j] - mx);
        accs[t][j] = pv;
        sum += pv;
      }
#pragma unroll
    for (int s = 1; s < 16; s <<= 1) sum += __shfl_xor(sum, s, 16);
    float inv = 1.0f / sum;
    int m = j + 8 * hh;
#pragma unroll
    for (int t = 0; t < 7; ++t)
      if (t * 16 < nkeys)
        my[m * 136 + t * 16 + lq] = (_Float16)(accs[t][j] * inv);
  }

  // ---- all async V transfers must have landed before reading the stage
  asm volatile("s_wait_asynccnt 0x0" ::: "memory");

  // ---- AV: attn (A, from LDS) x V (B, from LDS stage), K = 32 keys / WMMA
  v8f oacc = {};
#pragma unroll
  for (int kt = 0; kt < 4; ++kt) {
    if (kt * 32 < nkeys) {
      v16h av, vv;
#pragma unroll
      for (int e = 0; e < 16; ++e)
        av[e] = my[lq * 136 + kt * 32 + amapK(e, hh)];
#pragma unroll
      for (int e = 0; e < 16; ++e)
        vv[e] = vlds[(kt * 32 + hh * 16 + e) * HD + lq];
      oacc = __builtin_amdgcn_wmma_f32_16x16x32_f16(false, av, false, vv,
                                                    (short)0, oacc, false, false);
    }
  }

  // epilogue: row m = query, col lq = d; store f16 [b][pixel][C] for proj
#pragma unroll
  for (int j = 0; j < 8; ++j) {
    int m = j + 8 * hh;
    int hw = (qr0 + (m >> 2)) * WIMG + (qc0 + (m & 3));
    obuf[((size_t)b * HWPX + hw) * CCH + h * HD + lq] = (_Float16)oacc[j];
  }
}

// ---------------------------------------------------------------------------
// Kernel 3: out = einsum('bchw,oc->bohw', attn_out, proj_w) + proj_b, f32 out.
// Same tiling as kernel 1 but A (attn_out) is already f16 row-major [pixel][C].
// ---------------------------------------------------------------------------
__global__ __launch_bounds__(256) void proj_gemm(
    const _Float16* __restrict__ ib, const float* __restrict__ w,
    const float* __restrict__ bias, float* __restrict__ out) {
  const int wid = threadIdx.x >> 5, lane = threadIdx.x & 31;
  const int hh = lane >> 4, lq = lane & 15;
  int gid = blockIdx.x * 8 + wid;   // 8192 waves: 2048 M-tiles x 4 N-tiles
  int mt = gid >> 2, nt = gid & 3;
  int pix = mt * 16 + lq;
  const _Float16* arow = ib + (size_t)pix * CCH;

  v8f acc = {};
#pragma unroll
  for (int ks = 0; ks < 2; ++ks) {
    v16h a, bv;
#pragma unroll
    for (int e = 0; e < 16; ++e)
      a[e] = arow[ks * 32 + amapK(e, hh)];
    int n = nt * 16 + lq;
    const float4* wp = (const float4*)(w + (size_t)n * CCH + ks * 32 + hh * 16);
#pragma unroll
    for (int q = 0; q < 4; ++q) {
      float4 w4 = wp[q];
      bv[q * 4 + 0] = (_Float16)w4.x; bv[q * 4 + 1] = (_Float16)w4.y;
      bv[q * 4 + 2] = (_Float16)w4.z; bv[q * 4 + 3] = (_Float16)w4.w;
    }
    acc = __builtin_amdgcn_wmma_f32_16x16x32_f16(false, a, false, bv,
                                                 (short)0, acc, false, false);
  }
  int o = nt * 16 + lq;
  float bia = bias[o];
#pragma unroll
  for (int j = 0; j < 8; ++j) {
    int m = j + 8 * hh;
    int pj = mt * 16 + m;
    int bj = pj >> 14, hwj = pj & (HWPX - 1);
    out[((size_t)bj * CCH + o) * HWPX + hwj] = acc[j] + bia;
  }
}

// ---------------------------------------------------------------------------
extern "C" void kernel_launch(void* const* d_in, const int* in_sizes, int n_in,
                              void* d_out, int out_size, void* d_ws,
                              size_t ws_size, hipStream_t stream) {
  const float* x      = (const float*)d_in[0];
  const float* qkv_w  = (const float*)d_in[1];
  const float* qkv_b  = (const float*)d_in[2];
  const float* proj_w = (const float*)d_in[3];
  const float* proj_b = (const float*)d_in[4];
  const float* rbias  = (const float*)d_in[5];
  float* out = (float*)d_out;

  // workspace layout (f16): q | k | v (each B*NH*HW*HD = 2,097,152) | attn_out
  _Float16* qb = (_Float16*)d_ws;
  _Float16* kb = qb + 2097152;
  _Float16* vb = kb + 2097152;
  _Float16* ob = vb + 2097152;   // B*HW*C = 2,097,152 halfs; total 16 MB

  qkv_gemm<<<3072, 256, 0, stream>>>(x, qkv_w, qkv_b, qb, kb, vb);
  na_attn<<<2048, 128, 0, stream>>>(qb, kb, vb, rbias, ob);
  proj_gemm<<<1024, 256, 0, stream>>>(ob, proj_w, proj_b, out);
}
